// AttnBlock_2800318677413
// MI455X (gfx1250) — matrix operand order repaired and checked by reading, not compiled, not run on hardware
//
#include <hip/hip_runtime.h>

typedef __attribute__((ext_vector_type(16))) _Float16 v16h;
typedef __attribute__((ext_vector_type(8)))  _Float16 v8h;
typedef __attribute__((ext_vector_type(4)))  _Float16 v4h;
typedef __attribute__((ext_vector_type(8)))  float    v8f;

#define B_   4
#define C_   512
#define N_   4096
#define G_   32
#define CPG  16
#define EPSV 1e-6f

// ---------------------------------------------------------------------------
// Async global->LDS staging (CDNA5 GLOBAL_LOAD_ASYNC_TO_LDS_B128, ASYNCcnt).
// Probe-confirmed signature: (v4i AS1*, v4i AS3*, imm int, imm int),
// v4i = vector_size(16) int.
// ---------------------------------------------------------------------------
#if __has_builtin(__builtin_amdgcn_global_load_async_to_lds_b128)
#define ASYNC_LDS 1
typedef int v4i __attribute__((vector_size(16)));
typedef __attribute__((address_space(1))) v4i* gas_ptr_t;
typedef __attribute__((address_space(3))) v4i* las_ptr_t;
#endif

static __device__ __forceinline__ void stage16(const _Float16* g, _Float16* l) {
#ifdef ASYNC_LDS
  __builtin_amdgcn_global_load_async_to_lds_b128(
      (gas_ptr_t)(unsigned long long)g,
      (las_ptr_t)(unsigned int)(unsigned long long)l, 0, 0);
#else
  *(uint4*)l = *(const uint4*)g;
#endif
}

static __device__ __forceinline__ void stage_fence() {
#ifdef ASYNC_LDS
  asm volatile("s_wait_asynccnt 0" ::: "memory");
#endif
  __syncthreads();
}

// ---------------------------------------------------------------------------
// WMMA fragments as contiguous 16B loads (ISA 7.12.2 layouts):
// A (16x32): lane<16 row=lane K{0..7,16..23}; lane>=16 row=lane-16 K{8..15,24..31}
//   -> two contiguous 8-half chunks at kb and kb+16 of a row-major [row][k] tile.
// B (32x16): lane<16 col=lane K{0..7,16..23}; lane>=16 col=lane-16 K{8..15,24..31}
//   -> from a [col][k] row-major tile: two 8-half chunks at kb and kb+16.
// ---------------------------------------------------------------------------
static __device__ __forceinline__ v16h load_frag_a(const _Float16* t, int ld) {
  const int lane = threadIdx.x & 31;
  const int row  = lane & 15;
  const int kb   = (lane >> 4) << 3;                 // 0 or 8
  v8h lo = *(const v8h*)(t + row * ld + kb);
  v8h hi = *(const v8h*)(t + row * ld + kb + 16);
  return __builtin_shufflevector(lo, hi, 0, 1, 2, 3, 4, 5, 6, 7,
                                 8, 9, 10, 11, 12, 13, 14, 15);
}

static __device__ __forceinline__ v16h load_frag_bt(const _Float16* t, int ld) {
  // t: [col][k] row-major tile
  const int lane = threadIdx.x & 31;
  const int col  = lane & 15;
  const int kb   = (lane >> 4) << 3;                 // 0 or 8
  v8h lo = *(const v8h*)(t + col * ld + kb);
  v8h hi = *(const v8h*)(t + col * ld + kb + 16);
  return __builtin_shufflevector(lo, hi, 0, 1, 2, 3, 4, 5, 6, 7,
                                 8, 9, 10, 11, 12, 13, 14, 15);
}

static __device__ __forceinline__ v8f wmma_f16(v16h a, v16h b, v8f c) {
  return __builtin_amdgcn_wmma_f32_16x16x32_f16(false, a, false, b,
                                                (short)0, c, false, false);
}

// ---------------------------------------------------------------------------
// Kernel 1: GroupNorm.  x:[B][C][N] f32 -> hn:[B][N][C] f16.
// ---------------------------------------------------------------------------
__global__ __launch_bounds__(256) void groupnorm_kernel(
    const float* __restrict__ x, const float* __restrict__ gamma,
    const float* __restrict__ beta, _Float16* __restrict__ hn) {
  __shared__ float s_sum[256], s_sq[256];
  __shared__ float s_g[CPG], s_b[CPG];
  __shared__ float s_mean, s_inv;
  const int tid = threadIdx.x;
  const int bb  = blockIdx.x / G_;
  const int g   = blockIdx.x % G_;
  const size_t base = ((size_t)bb * C_ + (size_t)g * CPG) * N_;
  const float4* x4 = (const float4*)(x + base);

  float sum = 0.f, sq = 0.f;
#pragma unroll 4
  for (int i = 0; i < 64; ++i) {
    float4 v = x4[tid + 256 * i];
    sum += v.x + v.y + v.z + v.w;
    sq  += v.x * v.x + v.y * v.y + v.z * v.z + v.w * v.w;
  }
  s_sum[tid] = sum;
  s_sq[tid]  = sq;
  if (tid < CPG) { s_g[tid] = gamma[g * CPG + tid]; s_b[tid] = beta[g * CPG + tid]; }
  __syncthreads();
  for (int s = 128; s > 0; s >>= 1) {
    if (tid < s) { s_sum[tid] += s_sum[tid + s]; s_sq[tid] += s_sq[tid + s]; }
    __syncthreads();
  }
  if (tid == 0) {
    float mean = s_sum[0] * (1.f / (CPG * N_));
    float var  = s_sq[0] * (1.f / (CPG * N_)) - mean * mean;
    s_mean = mean;
    s_inv  = rsqrtf(var + EPSV);
  }
  __syncthreads();
  const float mean = s_mean, inv = s_inv;
#pragma unroll 2
  for (int i = 0; i < 64; ++i) {
    int e4 = tid + 256 * i;
    float4 v = x4[e4];
    float vv[4] = {v.x, v.y, v.z, v.w};
#pragma unroll
    for (int q = 0; q < 4; ++q) {
      int e  = e4 * 4 + q;
      int cl = e >> 12;
      int n  = e & (N_ - 1);
      float val = (vv[q] - mean) * inv * s_g[cl] + s_b[cl];
      hn[((size_t)bb * N_ + n) * C_ + g * CPG + cl] = (_Float16)val;
    }
  }
}

// ---------------------------------------------------------------------------
// Kernel 2: 1x1 conv as WMMA GEMM.  out[n,o] = sum_c A[n,c]*W[o,c] + bias[o]
// Macro-tile 128(n) x 64(o), 8 waves, wave = 2x2 tiles.
// MODE 0: f16 [B][N][O] (q,k); MODE 1: f16 [B][O][N] (v);
// MODE 2: f32 [B][O][N] + residual (final projection).
// ---------------------------------------------------------------------------
template <int MODE>
__global__ __launch_bounds__(256) void proj_kernel(
    const _Float16* __restrict__ A, const float* __restrict__ W,
    const float* __restrict__ bias, _Float16* __restrict__ outh,
    float* __restrict__ outf, const float* __restrict__ xres) {
  __shared__ __align__(16) _Float16 lds_a[128 * 40];  // [n][c] tile
  __shared__ __align__(16) _Float16 lds_b[64 * 40];   // [o][c] tile (B as [col][k])
  const int tid   = threadIdx.x;
  const int lane  = tid & 31;
  const int wv    = tid >> 5;
  const int bb    = blockIdx.z;
  const int n0    = blockIdx.x * 128;
  const int o0    = blockIdx.y * 64;
  const int rbase = (wv & 3) * 32;
  const int cbase = (wv >> 2) * 32;
  const _Float16* Ab = A + ((size_t)bb * N_ + n0) * C_;

  v8f acc[2][2] = {};
  for (int c0 = 0; c0 < C_; c0 += 32) {
    __syncthreads();
    // stage A panel: 128 rows x 32 halfs (async b128 -> LDS)
#pragma unroll
    for (int it = 0; it < 2; ++it) {
      int u = tid + 256 * it;                       // 0..511
      int row = u >> 2, seg = u & 3;
      stage16(Ab + (size_t)row * C_ + c0 + seg * 8, &lds_a[row * 40 + seg * 8]);
    }
    // stage W panel: 64 (o) x 32 (c), f32 -> f16 convert, same [o][c] order
#pragma unroll
    for (int it = 0; it < 2; ++it) {
      int u = tid + 256 * it;                       // 0..511
      int row = u >> 3, seg = u & 7;
      float4 w4 = *(const float4*)(W + (size_t)(o0 + row) * C_ + c0 + seg * 4);
      v4h h4 = {(_Float16)w4.x, (_Float16)w4.y, (_Float16)w4.z, (_Float16)w4.w};
      *(v4h*)&lds_b[row * 40 + seg * 4] = h4;
    }
    stage_fence();
    v16h a0 = load_frag_a(&lds_a[(rbase + 0) * 40], 40);
    v16h a1 = load_frag_a(&lds_a[(rbase + 16) * 40], 40);
    v16h b0 = load_frag_bt(&lds_b[(cbase + 0) * 40], 40);
    v16h b1 = load_frag_bt(&lds_b[(cbase + 16) * 40], 40);
    acc[0][0] = wmma_f16(a0, b0, acc[0][0]);
    acc[0][1] = wmma_f16(a0, b1, acc[0][1]);
    acc[1][0] = wmma_f16(a1, b0, acc[1][0]);
    acc[1][1] = wmma_f16(a1, b1, acc[1][1]);
  }
  const int hi = lane >> 4;
#pragma unroll
  for (int ti = 0; ti < 2; ++ti)
#pragma unroll
    for (int tj = 0; tj < 2; ++tj) {
      int o = o0 + cbase + 16 * tj + (lane & 15);
      float bv = bias[o];
#pragma unroll
      for (int r = 0; r < 8; ++r) {
        int n = n0 + rbase + 16 * ti + r + 8 * hi;
        float val = acc[ti][tj][r] + bv;
        if (MODE == 0) {
          outh[((size_t)bb * N_ + n) * C_ + o] = (_Float16)val;
        } else if (MODE == 1) {
          outh[((size_t)bb * C_ + o) * N_ + n] = (_Float16)val;
        } else {
          size_t idx = ((size_t)bb * C_ + o) * N_ + n;
          outf[idx] = xres[idx] + val;
        }
      }
    }
}

// ---------------------------------------------------------------------------
// Kernel 3: flash attention.  8 waves x 16 query rows; channel chunk 256
// (grid.y = 2) to halve K/V panel re-reads -- Q/K/V all sit in the 192MB L2,
// so re-read traffic, not HBM, is the only non-WMMA bound.  acc = 16 v8f
// (128 VGPRs/lane); occupancy is irrelevant for an XDL-bound loop, and the
// softmax VALU co-executes under the WMMA shadow.
// ---------------------------------------------------------------------------
__global__ __launch_bounds__(256) void attn_kernel(
    const _Float16* __restrict__ q, const _Float16* __restrict__ k,
    const _Float16* __restrict__ v, _Float16* __restrict__ o) {
  __shared__ __align__(16) _Float16 lds_kt[32 * 520];   // [j][c] K panel
  __shared__ __align__(16) _Float16 lds_v[256 * 40];    // [c][j] V panel
  __shared__ __align__(16) _Float16 lds_p[8 * 16 * 40]; // per-wave P tiles
  const int tid  = threadIdx.x;
  const int lane = tid & 31;
  const int wv   = tid >> 5;
  const int bb   = blockIdx.z;
  const int cb   = blockIdx.y * 256;
  const int i0   = blockIdx.x * 128 + wv * 16;
  const int hi   = lane >> 4;
  const float scale = 0.04419417382f;                   // 512^-0.5

  const _Float16* Qb = q + ((size_t)bb * N_ + i0) * C_;
  const _Float16* Kb = k + (size_t)bb * N_ * C_;
  const _Float16* Vb = v + (size_t)bb * C_ * N_;
  _Float16* Pw = &lds_p[wv * 16 * 40];

  v8f acc[16] = {};
  float mrow[8], lrow[8];
#pragma unroll
  for (int r = 0; r < 8; ++r) { mrow[r] = -1e30f; lrow[r] = 0.f; }

  for (int j0 = 0; j0 < N_; j0 += 32) {
    __syncthreads();
    // stage K panel: 32 j-rows x 512 c (async b128 -> LDS)
#pragma unroll
    for (int it = 0; it < 8; ++it) {
      int u = tid + 256 * it;                        // 0..2047
      int jj = u >> 6, seg = u & 63;
      stage16(Kb + (size_t)(j0 + jj) * C_ + seg * 8, &lds_kt[jj * 520 + seg * 8]);
    }
    // stage V panel: 256 c-rows x 32 j
#pragma unroll
    for (int it = 0; it < 4; ++it) {
      int u = tid + 256 * it;                        // 0..1023
      int cc = u >> 2, seg = u & 3;
      stage16(Vb + (size_t)(cb + cc) * N_ + j0 + seg * 8, &lds_v[cc * 40 + seg * 8]);
    }
    if (j0 + 32 < N_)
      __builtin_prefetch(Kb + (size_t)(j0 + 32 + (tid >> 3)) * C_ + (tid & 7) * 64, 0, 0);
    stage_fence();

    // S = Q * K^T : two 16x16 tiles, full C=512 contraction
    v8f s0 = {}, s1 = {};
#pragma unroll 4
    for (int c0 = 0; c0 < C_; c0 += 32) {
      v16h aq = load_frag_a(Qb + c0, C_);
      v16h b0 = load_frag_bt(&lds_kt[0 * 520 + c0], 520);
      v16h b1 = load_frag_bt(&lds_kt[16 * 520 + c0], 520);
      s0 = wmma_f16(aq, b0, s0);
      s1 = wmma_f16(aq, b1, s1);
    }

    // online softmax (16-lane groups share a physical row)
#pragma unroll
    for (int r = 0; r < 8; ++r) {
      float x0 = s0[r] * scale;
      float x1 = s1[r] * scale;
      float mx = fmaxf(x0, x1);
#pragma unroll
      for (int off = 1; off < 16; off <<= 1)
        mx = fmaxf(mx, __shfl_xor(mx, off, 32));
      float mnew = fmaxf(mrow[r], mx);
      float corr = __expf(mrow[r] - mnew);
      float p0 = __expf(x0 - mnew);
      float p1 = __expf(x1 - mnew);
      float ps = p0 + p1;
#pragma unroll
      for (int off = 1; off < 16; off <<= 1)
        ps += __shfl_xor(ps, off, 32);
      lrow[r] = lrow[r] * corr + ps;
      mrow[r] = mnew;
#pragma unroll
      for (int cc = 0; cc < 16; ++cc) acc[cc][r] *= corr;
      int row = r + 8 * hi;
      int col = lane & 15;
      Pw[row * 40 + col]      = (_Float16)p0;
      Pw[row * 40 + col + 16] = (_Float16)p1;
    }

    // O += P * V  (K=32 contraction over this key step)
    v16h ap = load_frag_a(Pw, 40);
#pragma unroll
    for (int cc = 0; cc < 16; ++cc) {
      v16h bv = load_frag_bt(&lds_v[(cc * 16) * 40], 40);
      acc[cc] = wmma_f16(ap, bv, acc[cc]);
    }
  }

#pragma unroll
  for (int r = 0; r < 8; ++r) {
    float oinv = 1.f / lrow[r];
    int n = i0 + r + 8 * hi;
#pragma unroll
    for (int cc = 0; cc < 16; ++cc) {
      int c = cb + cc * 16 + (lane & 15);
      o[((size_t)bb * N_ + n) * C_ + c] = (_Float16)(acc[cc][r] * oinv);
    }
  }
}

// ---------------------------------------------------------------------------
extern "C" void kernel_launch(void* const* d_in, const int* in_sizes, int n_in,
                              void* d_out, int out_size, void* d_ws,
                              size_t ws_size, hipStream_t stream) {
  const float* x   = (const float*)d_in[0];
  const float* gs  = (const float*)d_in[1];
  const float* gb  = (const float*)d_in[2];
  const float* wq  = (const float*)d_in[3];
  const float* bq  = (const float*)d_in[4];
  const float* wk  = (const float*)d_in[5];
  const float* bk  = (const float*)d_in[6];
  const float* wvp = (const float*)d_in[7];
  const float* bv  = (const float*)d_in[8];
  const float* wo  = (const float*)d_in[9];
  const float* bo  = (const float*)d_in[10];
  float* out = (float*)d_out;

  const size_t BUF = (size_t)B_ * N_ * C_;
  _Float16* hn = (_Float16*)d_ws;
  _Float16* qb = hn + BUF;
  _Float16* kb = qb + BUF;
  _Float16* vb = kb + BUF;
  _Float16* ob = vb + BUF;

  groupnorm_kernel<<<B_ * G_, 256, 0, stream>>>(x, gs, gb, hn);

  dim3 pg(N_ / 128, C_ / 64, B_);
  proj_kernel<0><<<pg, 256, 0, stream>>>(hn, wq, bq, qb, nullptr, nullptr);
  proj_kernel<0><<<pg, 256, 0, stream>>>(hn, wk, bk, kb, nullptr, nullptr);
  proj_kernel<1><<<pg, 256, 0, stream>>>(hn, wvp, bv, vb, nullptr, nullptr);

  attn_kernel<<<dim3(N_ / 128, C_ / 256, B_), 256, 0, stream>>>(qb, kb, vb, ob);

  proj_kernel<2><<<pg, 256, 0, stream>>>(ob, wo, bo, nullptr, out, x);

  (void)in_sizes; (void)n_in; (void)out_size; (void)ws_size;
}
